// Linear_22196390985759
// MI455X (gfx1250) — compile-verified, matrix operand order checked by reading, not run on hardware
//
#include <hip/hip_runtime.h>
#include <hip/hip_bf16.h>

typedef __bf16 bf16_t;
typedef __attribute__((ext_vector_type(16))) __bf16 v16bf;
typedef __attribute__((ext_vector_type(8)))  float  v8f;

#define HIDDEN 256
#define KDIM   512
#define EDGES_PER_BLOCK 128
#define KT_STEPS 16   // 512 / 32
#define NT_TILES 16   // 256 / 16

#if defined(__has_builtin)
#if __has_builtin(__builtin_amdgcn_global_load_async_to_lds_b128)
#define HAVE_ASYNC_LDS 1
#endif
#endif
#ifndef HAVE_ASYNC_LDS
#define HAVE_ASYNC_LDS 0
#endif

// native-vector pointer types matching the async-to-LDS builtin's signature:
//   param0: __attribute__((vector_size(16))) int  AS(1)*   (global src)
//   param1: same pointee, AS(3)                            (LDS dst)
typedef int nat_int4 __attribute__((vector_size(4 * sizeof(int))));
typedef __attribute__((address_space(1))) nat_int4 gas_int4;
typedef __attribute__((address_space(3))) nat_int4 las_int4;

// ---------------------------------------------------------------------------
// Pre-kernel 1: z (f32) -> bf16, row-major [N_NODES, 256]
// ---------------------------------------------------------------------------
__global__ void convert_z_bf16(const float* __restrict__ z,
                               bf16_t* __restrict__ zb, int n) {
  int i = (blockIdx.x * blockDim.x + threadIdx.x) * 4;
  if (i + 3 < n) {
    float4 f = *(const float4*)(z + i);
    zb[i + 0] = (bf16_t)f.x;
    zb[i + 1] = (bf16_t)f.y;
    zb[i + 2] = (bf16_t)f.z;
    zb[i + 3] = (bf16_t)f.w;
  } else {
    for (int j = i; j < n; ++j) zb[j] = (bf16_t)z[j];
  }
}

// ---------------------------------------------------------------------------
// Pre-kernel 2: repack W1 (f32 [512,256] row-major) into bf16 WMMA B-fragments.
//   lanes 0-15 : N = lane,    K = kt*32 + {0..7, 16..23}
//   lanes 16-31: N = lane-16, K = kt*32 + {8..15, 24..31}
// ---------------------------------------------------------------------------
__global__ void repack_w1(const float* __restrict__ W1,
                          bf16_t* __restrict__ w1t) {
  int t    = blockIdx.x * blockDim.x + threadIdx.x;  // 0..8191
  int tile = t >> 5;                                 // kt*16 + nt
  int lane = t & 31;
  int kt   = tile >> 4;
  int nt   = tile & 15;
  int n     = nt * 16 + (lane & 15);
  int kbase = kt * 32 + ((lane & 16) ? 8 : 0);
  bf16_t* dst = w1t + tile * 512 + lane * 16;
#pragma unroll
  for (int j = 0; j < 16; ++j) {
    int k = kbase + ((j < 8) ? j : (j + 8));
    dst[j] = (bf16_t)W1[k * HIDDEN + n];
  }
}

// ---------------------------------------------------------------------------
// Async LDS staging helpers (CDNA5 GLOBAL_LOAD_ASYNC_TO_LDS + ASYNCcnt)
// ---------------------------------------------------------------------------
__device__ __forceinline__ void stage_slab(const uint4* __restrict__ g,
                                           uint4* l, int tid) {
#if HAVE_ASYNC_LDS
#pragma unroll
  for (int i = 0; i < 4; ++i)
    __builtin_amdgcn_global_load_async_to_lds_b128(
        (gas_int4*)(g + tid + i * 256),
        (las_int4*)(l + tid + i * 256), 0, 0);
#else
#pragma unroll
  for (int i = 0; i < 4; ++i) l[tid + i * 256] = g[tid + i * 256];
#endif
}

template <int N>
__device__ __forceinline__ void wait_async() {
#if HAVE_ASYNC_LDS
#if __has_builtin(__builtin_amdgcn_s_wait_asynccnt)
  __builtin_amdgcn_s_wait_asynccnt((short)N);
#else
  asm volatile("s_wait_asynccnt %0" ::"i"(N) : "memory");
#endif
#endif
}

// ---------------------------------------------------------------------------
// Main kernel: 8 waves/block, 16 edges/wave, full N=256 per wave.
// ---------------------------------------------------------------------------
union AFrag { uint4 u[2]; v16bf v; };

__global__ __launch_bounds__(256) void edge_mlp_kernel(
    const bf16_t* __restrict__ zb,
    const bf16_t* __restrict__ w1t,
    const int*    __restrict__ src_idx,
    const int*    __restrict__ dst_idx,
    const float*  __restrict__ b1,
    const float*  __restrict__ W2,
    const float*  __restrict__ b2,
    float*        __restrict__ out) {
  __shared__ uint4 ldsB[2][NT_TILES * 64];  // 2 x 16 KB ping-pong slabs

  const int tid  = threadIdx.x;
  const int lane = tid & 31;
  const int wave = tid >> 5;
  const int m    = lane & 15;           // row within this wave's 16 edges
  const int hi   = (lane >> 4) & 1;     // upper lane half
  const int kkoff = hi ? 8 : 0;

  const int eBase = blockIdx.x * EDGES_PER_BLOCK + wave * 16;
  const int e     = eBase + m;

  const bf16_t* srcRow = zb + (size_t)src_idx[e] * HIDDEN;
  const bf16_t* dstRow = zb + (size_t)dst_idx[e] * HIDDEN;
  const uint4*  w1t4   = (const uint4*)w1t;

  v8f acc[NT_TILES];
#pragma unroll
  for (int i = 0; i < NT_TILES; ++i) acc[i] = (v8f){0, 0, 0, 0, 0, 0, 0, 0};

  // ---- prologue: slab 0 in flight, A fragment 0 in flight
  stage_slab(w1t4, &ldsB[0][0], tid);
  AFrag A[2];
  {
    const bf16_t* row = srcRow;
    A[0].u[0] = *(const uint4*)(row + kkoff);
    A[0].u[1] = *(const uint4*)(row + kkoff + 16);
  }

  for (int kt = 0; kt < KT_STEPS; ++kt) {
    const int cur = kt & 1;

    if (kt + 1 < KT_STEPS) {
      // issue next slab + next A fragment, then drain only the current slab
      stage_slab(w1t4 + (kt + 1) * 1024, &ldsB[cur ^ 1][0], tid);
      const bf16_t* row = ((kt + 1) < 8) ? srcRow : dstRow;
      const int kk = (((kt + 1) & 7) * 32) + kkoff;
      A[cur ^ 1].u[0] = *(const uint4*)(row + kk);
      A[cur ^ 1].u[1] = *(const uint4*)(row + kk + 16);
      wait_async<4>();
    } else {
      wait_async<0>();
    }
    __syncthreads();  // slab `cur` fully resident for all waves

    const uint4* bufc = &ldsB[cur][0];
    union BFrag { uint4 u[2]; v16bf v; } B[2];
    B[0].u[0] = bufc[lane * 2];
    B[0].u[1] = bufc[lane * 2 + 1];
#pragma unroll
    for (int nt = 0; nt < NT_TILES; ++nt) {
      if (nt + 1 < NT_TILES) {  // prefetch next B fragment into other regs
        B[(nt + 1) & 1].u[0] = bufc[(nt + 1) * 64 + lane * 2];
        B[(nt + 1) & 1].u[1] = bufc[(nt + 1) * 64 + lane * 2 + 1];
      }
      acc[nt] = __builtin_amdgcn_wmma_f32_16x16x32_bf16(
          false, A[cur].v, false, B[nt & 1].v, (short)0, acc[nt], false, false);
    }
    __syncthreads();  // protect ldsB[cur] before it is refilled at kt+1
  }

  // ---- fused epilogue: relu(h + b1) dot W2, reduced over N in-register.
  // C layout: acc[nt] vgpr r, lane l: M = r + hi*8, N = nt*16 + (l&15).
  const int nlane = lane & 15;
  float s[8];
#pragma unroll
  for (int r = 0; r < 8; ++r) s[r] = 0.f;
#pragma unroll
  for (int nt = 0; nt < NT_TILES; ++nt) {
    const int n   = nt * 16 + nlane;
    const float b1n = b1[n];
    const float w2n = W2[n];
#pragma unroll
    for (int r = 0; r < 8; ++r) {
      float h = acc[nt][r] + b1n;
      h = h > 0.f ? h : 0.f;
      s[r] += h * w2n;
    }
  }
#pragma unroll
  for (int r = 0; r < 8; ++r) {  // reduce across each 16-lane half
    float v = s[r];
    v += __shfl_xor(v, 1);
    v += __shfl_xor(v, 2);
    v += __shfl_xor(v, 4);
    v += __shfl_xor(v, 8);
    s[r] = v;
  }
  if (nlane == 0) {
    const float bias2 = b2[0];
#pragma unroll
    for (int r = 0; r < 8; ++r) out[eBase + r + hi * 8] = s[r] + bias2;
  }
}

// ---------------------------------------------------------------------------
extern "C" void kernel_launch(void* const* d_in, const int* in_sizes, int n_in,
                              void* d_out, int out_size, void* d_ws,
                              size_t ws_size, hipStream_t stream) {
  (void)n_in; (void)ws_size; (void)out_size;
  const float* z   = (const float*)d_in[0];
  const int*   src = (const int*)d_in[1];
  const int*   dst = (const int*)d_in[2];
  const float* W1  = (const float*)d_in[3];
  const float* b1  = (const float*)d_in[4];
  const float* W2  = (const float*)d_in[5];
  const float* b2  = (const float*)d_in[6];
  float* out = (float*)d_out;

  // workspace layout: [ z_bf16 | W1 fragment tiles ]
  const int zElems = in_sizes[0];                       // N_NODES * 256
  size_t zbBytes = ((size_t)zElems * 2 + 1023) & ~(size_t)1023;
  bf16_t* zb  = (bf16_t*)d_ws;
  bf16_t* w1t = (bf16_t*)((char*)d_ws + zbBytes);

  int zBlocks = (zElems / 4 + 255) / 256;
  convert_z_bf16<<<zBlocks, 256, 0, stream>>>(z, zb, zElems);
  repack_w1<<<32, 256, 0, stream>>>(W1, w1t);

  const int nEdges = in_sizes[1];                       // 524288
  const int blocks = nEdges / EDGES_PER_BLOCK;          // 4096
  edge_mlp_kernel<<<blocks, 256, 0, stream>>>(zb, w1t, src, dst, b1, W2, b2,
                                              out);
}
